// ETNNForecastingModel_88622355186343
// MI455X (gfx1250) — compile-verified, matrix operand order checked by reading, not run on hardware
//
#include <hip/hip_runtime.h>
#include <stdint.h>

typedef _Float16 v16h __attribute__((ext_vector_type(16)));
typedef _Float16 v8h  __attribute__((ext_vector_type(8)));
typedef float    v8f  __attribute__((ext_vector_type(8)));
typedef unsigned int u32x4 __attribute__((ext_vector_type(4)));
typedef int i32x4 __attribute__((ext_vector_type(4)));
typedef int i32x8 __attribute__((ext_vector_type(8)));
typedef int ivec4 __attribute__((vector_size(16)));   // gcc-style, matches builtin proto

#define BB 32
#define FF 128
#define HH 256
#define NN 2048
#define LL 2
#define EE 262144
#define NNODE (BB * NN)      // 65536
#define DEG_F 4.0f           // E // (B*N)
#define MT 32                // edges/nodes per block iteration (2 WMMA M-tiles)

// -------- TDM availability (5-arg on clang-22 / 6-arg on clang-23) -------------
#if defined(__gfx1250__) && \
    __has_builtin(__builtin_amdgcn_tensor_load_to_lds) && \
    __has_builtin(__builtin_amdgcn_s_wait_tensorcnt)
#define USE_TDM 1
#else
#define USE_TDM 0
#endif

// -------- async global->LDS (guarded; fallback = regular ld/st staging) --------
#if defined(__gfx1250__) && \
    __has_builtin(__builtin_amdgcn_global_load_async_to_lds_b128) && \
    __has_builtin(__builtin_amdgcn_s_wait_asynccnt)
#define USE_ASYNC 1
#else
#define USE_ASYNC 0
#endif

#if USE_ASYNC
__device__ __forceinline__ void async_cp16(void* lds, const void* g) {
  __builtin_amdgcn_global_load_async_to_lds_b128(
      (__attribute__((address_space(1))) ivec4*)g,
      (__attribute__((address_space(3))) ivec4*)lds, 0, 0);
}
#endif

#if USE_TDM
// Issue one 2D TDM load: 256x256 f16 row-major tile, global -> LDS.
__device__ __forceinline__ void tdm_load_256x256_f16(const void* gsrc, void* ldst) {
  unsigned long long ga = (unsigned long long)(uintptr_t)gsrc;
  u32x4 g0;
  g0[0] = 1u;                                   // count=1, user descriptor
  g0[1] = (unsigned)(uintptr_t)ldst;            // LDS byte address
  g0[2] = (unsigned)ga;                         // global_addr[31:0]
  g0[3] = (unsigned)((ga >> 32) & 0x01ffffffu)  // global_addr[56:32]
        | (2u << 30);                           // type=2 (image)
  i32x8 g1;
  g1[0] = 1 << 16;             // data_size = 1 -> 2 bytes
  g1[1] = (int)(256u << 16);   // tensor_dim0 = 256 (bits 63:48)
  g1[2] = (int)(256u << 16);   // tensor_dim1 = 256 (bits 111:80 low half)
  g1[3] = (int)(256u << 16);   // tile_dim0   = 256 (bits 127:112)
  g1[4] = 256;                 // tile_dim1   = 256 (bits 143:128)
  g1[5] = 256;                 // tensor_dim0_stride = 256 (bits 207:160 low32)
  g1[6] = 0;
  g1[7] = 0;
  i32x4 z4 = {0, 0, 0, 0};     // groups 2/3 disabled (2D tensor)
#if __clang_major__ >= 23
  i32x8 z8 = {0, 0, 0, 0, 0, 0, 0, 0};
  __builtin_amdgcn_tensor_load_to_lds(g0, g1, z4, z4, z8, 0);
#else
  __builtin_amdgcn_tensor_load_to_lds(g0, g1, z4, z4, 0);
#endif
}
#endif

// -------- WMMA fragment loader -------------------------------------------------
// Row-major [rows][256] f16 buffer (LDS). ISA 7.12.2 16-bit A layout (B uses the
// same mapping because weights are stored pre-transposed [N][K]):
//   lane L, element i: k = k0 + (i&7) + (i<8?0:16) + (L>=16?8:0)
__device__ __forceinline__ v16h ld_frag(const _Float16* base, int row, int k0) {
  int lane = threadIdx.x & 31;
  int koff = k0 + ((lane & 16) >> 1);           // +8 for upper half-wave
  const v8h* p = (const v8h*)(base + row * HH + koff);
  v8h lo = p[0];
  v8h hi = p[2];                                 // +16 halves
  v16h r;
#pragma unroll
  for (int i = 0; i < 8; ++i) { r[i] = lo[i]; r[i + 8] = hi[i]; }
  return r;
}

// =========================== small prep kernels ================================

__global__ void prep_weights_kernel(const float* __restrict__ Wm, const float* __restrict__ Wu,
                                    const float* __restrict__ Wg1,
                                    _Float16* __restrict__ W1T, _Float16* __restrict__ W2T,
                                    _Float16* __restrict__ WU1T, _Float16* __restrict__ WU2T,
                                    float* __restrict__ wlast, float* __restrict__ wg1sum) {
  int i = blockIdx.x * 256 + threadIdx.x;         // up to LL*HH*HH = 131072
  if (i < LL * HH * HH) {
    int l = i / (HH * HH);
    int n = (i / HH) & (HH - 1);
    int k = i & (HH - 1);
    size_t mb = (size_t)l * 513 * HH;
    size_t ub = (size_t)l * 512 * HH;
    W1T[i]  = (_Float16)Wm[mb + (size_t)k * HH + n];
    W2T[i]  = (_Float16)Wm[mb + (size_t)(k + 256) * HH + n];
    WU1T[i] = (_Float16)Wu[ub + (size_t)k * HH + n];
    WU2T[i] = (_Float16)Wu[ub + (size_t)(k + 256) * HH + n];
  }
  if (i < LL * HH) wlast[i] = Wm[(size_t)(i >> 8) * 513 * HH + 512 * HH + (i & (HH - 1))];
  if (i < FF) wg1sum[i] = Wg1[i] + Wg1[FF + i] + Wg1[2 * FF + i];
}

__global__ void h0_kernel(const float* __restrict__ x, const float* __restrict__ W_in,
                          const float* __restrict__ b_in, float* __restrict__ h0) {
  int b = blockIdx.x, c = threadIdx.x;
  float acc = b_in[c];
  for (int f = 0; f < FF; ++f) acc += x[b * FF + f] * W_in[f * HH + c];
  h0[b * HH + c] = acc;
}

__global__ void meand_kernel(const float* __restrict__ positions, float* __restrict__ meand) {
  __shared__ float sp[NN * 3];
  int tid = threadIdx.x;
  for (int i = tid; i < NN * 3; i += 256) sp[i] = positions[i];
  __syncthreads();
  int n = blockIdx.x * 256 + tid;
  float px = sp[n * 3], py = sp[n * 3 + 1], pz = sp[n * 3 + 2];
  float sum = 0.0f;
  for (int j = 0; j < NN; ++j) {
    float dx = px - sp[j * 3], dy = py - sp[j * 3 + 1], dz = pz - sp[j * 3 + 2];
    float dsq = dx * dx + dy * dy + dz * dz;
    if (j != n) sum += sqrtf(dsq);
  }
  meand[n] = sum / (float)(NN - 1);
}

__global__ void gfeat_kernel(const float* __restrict__ meand, const float* __restrict__ wg1sum,
                             const float* __restrict__ bg1, const float* __restrict__ Wg2,
                             const float* __restrict__ bg2, float* __restrict__ g) {
  __shared__ float hid[FF];
  int n = blockIdx.x, tid = threadIdx.x;
  float md = meand[n];
  if (tid < FF) hid[tid] = fmaxf(md * wg1sum[tid] + bg1[tid], 0.0f);
  __syncthreads();
  float acc = bg2[tid];
  for (int j = 0; j < FF; ++j) acc += hid[j] * Wg2[j * HH + tid];
  g[(size_t)n * HH + tid] = acc;
}

__global__ void init_h_kernel(const float* __restrict__ h0, const float* __restrict__ g,
                              float* __restrict__ h, _Float16* __restrict__ h16) {
  size_t i = (size_t)blockIdx.x * 256 + threadIdx.x;   // NNODE*HH threads
  int row = (int)(i >> 8), c = (int)(i & (HH - 1));
  float v = h0[(row >> 11) * HH + c] + g[(size_t)(row & (NN - 1)) * HH + c];
  h[i] = v;
  h16[i] = (_Float16)v;
}

__global__ void init_pos_kernel(const float* __restrict__ positions, float* __restrict__ pos) {
  int i = blockIdx.x * 256 + threadIdx.x;
  if (i < NNODE * 3) {
    int row = i / 3, s = i - row * 3;
    pos[i] = positions[(row & (NN - 1)) * 3 + s];
  }
}

__global__ void pos_update_kernel(float* __restrict__ pos, const float* __restrict__ dpos) {
  int i = blockIdx.x * 256 + threadIdx.x;
  if (i < NNODE * 3) pos[i] += dpos[i] * (1.0f / DEG_F);
}

__global__ void out_kernel(const float* __restrict__ h, const float* __restrict__ W_out,
                           const float* __restrict__ b_out, float* __restrict__ out) {
  int lane = threadIdx.x & 31;
  int w = blockIdx.x * (blockDim.x >> 5) + (threadIdx.x >> 5);
  if (w >= NNODE) return;
  float acc = 0.0f;
  for (int c = lane; c < HH; c += 32) acc += h[(size_t)w * HH + c] * W_out[c];
  for (int off = 16; off; off >>= 1) acc += __shfl_xor(acc, off, 32);
  if (lane == 0) out[w] = acc + b_out[0];
}

// =========================== edge message kernel ===============================
// m = relu(h[src]@W1 + h[dst]@W2 + d*wlast + bm);  agg[dst]+=m;
// coef = m@Wp;  dpos[dst] += (pos[dst]-pos[src])*coef
__global__ void __launch_bounds__(128)
edge_msg_kernel(const _Float16* __restrict__ h16, const float* __restrict__ pos,
                const _Float16* __restrict__ W1T, const _Float16* __restrict__ W2T,
                const float* __restrict__ wlast, const float* __restrict__ bm,
                const float* __restrict__ wp,
                const int* __restrict__ src, const int* __restrict__ dst,
                float* __restrict__ agg, float* __restrict__ dpos) {
  extern __shared__ char smem[];
  _Float16* w1 = (_Float16*)smem;                    // 256x256 f16 (W1^T, [N][K])
  _Float16* w2 = w1 + HH * HH;                       // 256x256 f16
  _Float16* hs = w2 + HH * HH;                       // MT x 256 f16 gathered h[src]
  _Float16* hd = hs + MT * HH;                       // MT x 256 f16 gathered h[dst]
  float* s_wlast = (float*)(hd + MT * HH);
  float* s_bm    = s_wlast + HH;
  float* s_wp    = s_bm + HH;
  float* s_d     = s_wp + HH;                        // MT
  float* s_pd    = s_d + MT;                         // MT x 3 (pos[dst]-pos[src])
  float* s_coef  = s_pd + MT * 3;                    // MT
  int*   s_si    = (int*)(s_coef + MT);              // MT
  int*   s_di    = s_si + MT;                        // MT

  const int tid = threadIdx.x;
  const int lane = tid & 31;
  const int wv = tid >> 5;
  const int am = lane & 15;
  const int rbase = (lane & 16) >> 1;                // 0 or 8

  // ---- stage weights once per block ----
#if USE_TDM
  if (wv == 0) {
    tdm_load_256x256_f16(W1T, w1);
    tdm_load_256x256_f16(W2T, w2);
    __builtin_amdgcn_s_wait_tensorcnt(0);
  }
#else
  {
    const uint4* g1p = (const uint4*)W1T;
    const uint4* g2p = (const uint4*)W2T;
    uint4* l1 = (uint4*)w1;
    uint4* l2 = (uint4*)w2;
    for (int i = tid; i < HH * HH / 8; i += 128) { l1[i] = g1p[i]; l2[i] = g2p[i]; }
  }
#endif
  for (int i = tid; i < HH; i += 128) {
    s_wlast[i] = wlast[i]; s_bm[i] = bm[i]; s_wp[i] = wp[i];
  }
  __syncthreads();

  const int ntiles = EE / MT;
  for (int tile = blockIdx.x; tile < ntiles; tile += (int)gridDim.x) {
    __syncthreads();   // previous iteration fully consumed staging
    __builtin_prefetch(&src[(tile + (int)gridDim.x) * MT & (EE - 1)], 0, 0);
    if (tid < MT) {
      int e = tile * MT + tid;
      int s = src[e], d = dst[e];
      s_si[tid] = s; s_di[tid] = d;
      float dx = pos[(size_t)s * 3 + 0] - pos[(size_t)d * 3 + 0];
      float dy = pos[(size_t)s * 3 + 1] - pos[(size_t)d * 3 + 1];
      float dz = pos[(size_t)s * 3 + 2] - pos[(size_t)d * 3 + 2];
      s_d[tid] = sqrtf(dx * dx + dy * dy + dz * dz + 1e-12f);
      s_pd[tid * 3 + 0] = -dx;   // pos[dst]-pos[src]
      s_pd[tid * 3 + 1] = -dy;
      s_pd[tid * 3 + 2] = -dz;
      s_coef[tid] = 0.0f;
    }
    __syncthreads();
    // gather MT src and MT dst rows into LDS (MT rows x 32 uint4 each)
    {
      uint4* ls = (uint4*)hs;
      uint4* ld = (uint4*)hd;
#pragma unroll
      for (int it = 0; it < MT / 4; ++it) {
        int v = tid + it * 128;               // 0 .. MT*32-1
        int r = v >> 5, c = v & 31;
        const uint4* gs = (const uint4*)(h16 + (size_t)s_si[r] * HH) + c;
        const uint4* gd = (const uint4*)(h16 + (size_t)s_di[r] * HH) + c;
#if USE_ASYNC
        async_cp16(ls + v, gs);
        async_cp16(ld + v, gd);
#else
        ls[v] = *gs;
        ld[v] = *gd;
#endif
      }
#if USE_ASYNC
      __builtin_amdgcn_s_wait_asynccnt(0);
#endif
    }
    __syncthreads();

    v8f acc[2][4] = {};
#pragma unroll
    for (int ks = 0; ks < 8; ++ks) {
      int k0 = ks * 32;
      v16h as0 = ld_frag(hs, am, k0);
      v16h as1 = ld_frag(hs, 16 + am, k0);
      v16h ad0 = ld_frag(hd, am, k0);
      v16h ad1 = ld_frag(hd, 16 + am, k0);
#pragma unroll
      for (int t = 0; t < 4; ++t) {
        int nrow = (wv * 4 + t) * 16 + am;
        v16h b1 = ld_frag(w1, nrow, k0);
        v16h b2 = ld_frag(w2, nrow, k0);
        acc[0][t] = __builtin_amdgcn_wmma_f32_16x16x32_f16(false, as0, false, b1,
                                                           (short)0, acc[0][t], false, false);
        acc[1][t] = __builtin_amdgcn_wmma_f32_16x16x32_f16(false, as1, false, b1,
                                                           (short)0, acc[1][t], false, false);
        acc[0][t] = __builtin_amdgcn_wmma_f32_16x16x32_f16(false, ad0, false, b2,
                                                           (short)0, acc[0][t], false, false);
        acc[1][t] = __builtin_amdgcn_wmma_f32_16x16x32_f16(false, ad1, false, b2,
                                                           (short)0, acc[1][t], false, false);
      }
    }

    // epilogue: relu, agg scatter-add, coef reduction
#pragma unroll
    for (int mt = 0; mt < 2; ++mt) {
#pragma unroll
      for (int t = 0; t < 4; ++t) {
        int col = (wv * 4 + t) * 16 + am;
        float wl = s_wlast[col], bb = s_bm[col], wpc = s_wp[col];
#pragma unroll
        for (int r = 0; r < 8; ++r) {
          int er = mt * 16 + r + rbase;
          float m = acc[mt][t][r] + s_d[er] * wl + bb;
          m = m > 0.0f ? m : 0.0f;
          atomicAdd(&agg[(size_t)s_di[er] * HH + col], m);
          float c = m * wpc;
          c += __shfl_xor(c, 1, 32);
          c += __shfl_xor(c, 2, 32);
          c += __shfl_xor(c, 4, 32);
          c += __shfl_xor(c, 8, 32);          // sum over the 16-lane half
          if (am == 0) atomicAdd(&s_coef[er], c);
        }
      }
    }
    __syncthreads();
    if (tid < MT) {
      float c = s_coef[tid];
      int d = s_di[tid];
      atomicAdd(&dpos[(size_t)d * 3 + 0], s_pd[tid * 3 + 0] * c);
      atomicAdd(&dpos[(size_t)d * 3 + 1], s_pd[tid * 3 + 1] * c);
      atomicAdd(&dpos[(size_t)d * 3 + 2], s_pd[tid * 3 + 2] * c);
    }
  }
}

// =========================== node update kernel ================================
// h += relu(h@Wu1 + agg@Wu2 + bu); refresh f16 shadow (in place, rows disjoint)
__global__ void __launch_bounds__(128)
node_update_kernel(float* __restrict__ h, _Float16* __restrict__ h16,
                   const float* __restrict__ agg,
                   const _Float16* __restrict__ WU1T, const _Float16* __restrict__ WU2T,
                   const float* __restrict__ bu) {
  extern __shared__ char smem[];
  _Float16* w1 = (_Float16*)smem;
  _Float16* w2 = w1 + HH * HH;
  _Float16* hh = w2 + HH * HH;                       // MT x 256 f16
  _Float16* aa = hh + MT * HH;                       // MT x 256 f16 (converted agg)
  float* s_bu = (float*)(aa + MT * HH);

  const int tid = threadIdx.x;
  const int lane = tid & 31;
  const int wv = tid >> 5;
  const int am = lane & 15;
  const int rbase = (lane & 16) >> 1;

#if USE_TDM
  if (wv == 0) {
    tdm_load_256x256_f16(WU1T, w1);
    tdm_load_256x256_f16(WU2T, w2);
    __builtin_amdgcn_s_wait_tensorcnt(0);
  }
#else
  {
    const uint4* g1p = (const uint4*)WU1T;
    const uint4* g2p = (const uint4*)WU2T;
    uint4* l1 = (uint4*)w1;
    uint4* l2 = (uint4*)w2;
    for (int i = tid; i < HH * HH / 8; i += 128) { l1[i] = g1p[i]; l2[i] = g2p[i]; }
  }
#endif
  for (int i = tid; i < HH; i += 128) s_bu[i] = bu[i];
  __syncthreads();

  const int ntiles = NNODE / MT;
  for (int tile = blockIdx.x; tile < ntiles; tile += (int)gridDim.x) {
    __syncthreads();
    int row0 = tile * MT;
    {
      uint4* lh = (uint4*)hh;
#pragma unroll
      for (int it = 0; it < MT / 4; ++it) {
        int v = tid + it * 128;               // 0 .. MT*32-1
        int r = v >> 5, c = v & 31;
        const uint4* gp = (const uint4*)(h16 + (size_t)(row0 + r) * HH) + c;
#if USE_ASYNC
        async_cp16(lh + v, gp);
#else
        lh[v] = *gp;
#endif
      }
#pragma unroll
      for (int it = 0; it < MT / 2; ++it) {
        int v = tid + it * 128;               // 0 .. MT*64-1 (MT rows x 64 float4)
        int r = v >> 6, c4 = v & 63;
        float4 f = ((const float4*)(agg + (size_t)(row0 + r) * HH))[c4];
        _Float16* dp = aa + r * HH + (c4 << 2);
        dp[0] = (_Float16)f.x; dp[1] = (_Float16)f.y;
        dp[2] = (_Float16)f.z; dp[3] = (_Float16)f.w;
      }
#if USE_ASYNC
      __builtin_amdgcn_s_wait_asynccnt(0);
#endif
    }
    __syncthreads();

    v8f acc[2][4] = {};
#pragma unroll
    for (int ks = 0; ks < 8; ++ks) {
      int k0 = ks * 32;
      v16h a10 = ld_frag(hh, am, k0);
      v16h a11 = ld_frag(hh, 16 + am, k0);
      v16h a20 = ld_frag(aa, am, k0);
      v16h a21 = ld_frag(aa, 16 + am, k0);
#pragma unroll
      for (int t = 0; t < 4; ++t) {
        int nrow = (wv * 4 + t) * 16 + am;
        v16h b1 = ld_frag(w1, nrow, k0);
        v16h b2 = ld_frag(w2, nrow, k0);
        acc[0][t] = __builtin_amdgcn_wmma_f32_16x16x32_f16(false, a10, false, b1,
                                                           (short)0, acc[0][t], false, false);
        acc[1][t] = __builtin_amdgcn_wmma_f32_16x16x32_f16(false, a11, false, b1,
                                                           (short)0, acc[1][t], false, false);
        acc[0][t] = __builtin_amdgcn_wmma_f32_16x16x32_f16(false, a20, false, b2,
                                                           (short)0, acc[0][t], false, false);
        acc[1][t] = __builtin_amdgcn_wmma_f32_16x16x32_f16(false, a21, false, b2,
                                                           (short)0, acc[1][t], false, false);
      }
    }
#pragma unroll
    for (int mt = 0; mt < 2; ++mt) {
#pragma unroll
      for (int t = 0; t < 4; ++t) {
        int col = (wv * 4 + t) * 16 + am;
        float bb = s_bu[col];
#pragma unroll
        for (int r = 0; r < 8; ++r) {
          int row = row0 + mt * 16 + r + rbase;
          float v = acc[mt][t][r] + bb;
          v = v > 0.0f ? v : 0.0f;
          float hn = h[(size_t)row * HH + col] + v;
          h[(size_t)row * HH + col] = hn;
          h16[(size_t)row * HH + col] = (_Float16)hn;
        }
      }
    }
  }
}

// =========================== launcher ==========================================
extern "C" void kernel_launch(void* const* d_in, const int* in_sizes, int n_in,
                              void* d_out, int out_size, void* d_ws, size_t ws_size,
                              hipStream_t stream) {
  const float* x         = (const float*)d_in[0];
  const float* positions = (const float*)d_in[1];
  const float* W_in      = (const float*)d_in[2];
  const float* b_in      = (const float*)d_in[3];
  const float* Wg1       = (const float*)d_in[4];
  const float* bg1       = (const float*)d_in[5];
  const float* Wg2       = (const float*)d_in[6];
  const float* bg2       = (const float*)d_in[7];
  const float* Wm        = (const float*)d_in[8];
  const float* bm        = (const float*)d_in[9];
  const float* Wu        = (const float*)d_in[10];
  const float* bu        = (const float*)d_in[11];
  const float* Wp        = (const float*)d_in[12];
  const float* W_out     = (const float*)d_in[13];
  const float* b_out     = (const float*)d_in[14];
  const int*   edge      = (const int*)d_in[15];
  const int* src = edge;
  const int* dst = edge + EE;

  char* ws = (char*)d_ws;
  size_t off = 0;
  auto alloc = [&](size_t bytes) -> void* {
    void* p = (void*)(ws + off);
    off += (bytes + 255) & ~(size_t)255;
    return p;
  };
  float*    h     = (float*)alloc((size_t)NNODE * HH * 4);
  _Float16* h16   = (_Float16*)alloc((size_t)NNODE * HH * 2);
  float*    agg   = (float*)alloc((size_t)NNODE * HH * 4);
  float*    pos   = (float*)alloc((size_t)NNODE * 3 * 4);
  float*    dpos  = (float*)alloc((size_t)NNODE * 3 * 4);
  float*    h0    = (float*)alloc((size_t)BB * HH * 4);
  float*    g     = (float*)alloc((size_t)NN * HH * 4);
  float*    meand = (float*)alloc((size_t)NN * 4);
  _Float16* W1T   = (_Float16*)alloc((size_t)LL * HH * HH * 2);
  _Float16* W2T   = (_Float16*)alloc((size_t)LL * HH * HH * 2);
  _Float16* WU1T  = (_Float16*)alloc((size_t)LL * HH * HH * 2);
  _Float16* WU2T  = (_Float16*)alloc((size_t)LL * HH * HH * 2);
  float*    wlast = (float*)alloc((size_t)LL * HH * 4);
  float*    wg1s  = (float*)alloc((size_t)FF * 4);
  (void)ws_size; (void)in_sizes; (void)n_in; (void)out_size;

  // ---- prologue ----
  prep_weights_kernel<<<(LL * HH * HH) / 256, 256, 0, stream>>>(
      Wm, Wu, Wg1, W1T, W2T, WU1T, WU2T, wlast, wg1s);
  h0_kernel<<<BB, HH, 0, stream>>>(x, W_in, b_in, h0);
  meand_kernel<<<NN / 256, 256, 0, stream>>>(positions, meand);
  gfeat_kernel<<<NN, HH, 0, stream>>>(meand, wg1s, bg1, Wg2, bg2, g);
  init_h_kernel<<<(NNODE * HH) / 256, 256, 0, stream>>>(h0, g, h, h16);
  init_pos_kernel<<<(NNODE * 3 + 255) / 256, 256, 0, stream>>>(positions, pos);

  // dynamic LDS: weights(256KB) + staging(32KB) + small tables
  size_t lds_d = (size_t)HH * HH * 2 * 2 + 2 * MT * HH * 2 + 3 * HH * 4
               + (MT + MT * 3 + MT) * 4 + 2 * MT * 4;
  size_t lds_e = (size_t)HH * HH * 2 * 2 + 2 * MT * HH * 2 + HH * 4;

  for (int l = 0; l < LL; ++l) {
    (void)hipMemsetAsync(agg, 0, (size_t)NNODE * HH * 4, stream);
    (void)hipMemsetAsync(dpos, 0, (size_t)NNODE * 3 * 4, stream);
    edge_msg_kernel<<<2048, 128, lds_d, stream>>>(
        h16, pos,
        W1T + (size_t)l * HH * HH, W2T + (size_t)l * HH * HH,
        wlast + (size_t)l * HH, bm + (size_t)l * HH, Wp + (size_t)l * HH,
        src, dst, agg, dpos);
    node_update_kernel<<<1024, 128, lds_e, stream>>>(
        h, h16, agg,
        WU1T + (size_t)l * HH * HH, WU2T + (size_t)l * HH * HH,
        bu + (size_t)l * HH);
    pos_update_kernel<<<(NNODE * 3 + 255) / 256, 256, 0, stream>>>(pos, dpos);
  }

  out_kernel<<<NNODE / 8, 256, 0, stream>>>(h, W_out, b_out, (float*)d_out);
}
